// DecoderHead_27049704031062
// MI455X (gfx1250) — matrix operand order repaired and checked by reading, not compiled, not run on hardware
//
#include <hip/hip_runtime.h>
#include <hip/hip_bf16.h>

typedef __bf16 bf16_t;
typedef __attribute__((ext_vector_type(16))) __bf16       v16bf;
typedef __attribute__((ext_vector_type(8)))  float        v8f;
typedef __attribute__((ext_vector_type(4)))  unsigned int v4u;
typedef __attribute__((ext_vector_type(8)))  int          v8i;
typedef __attribute__((ext_vector_type(4)))  int          v4i;

// Problem constants (match reference)
constexpr int Bc = 2, Tc = 2048, Dc = 1024, Hc = 16, DHc = 64;
constexpr float SCALE = 0.03125f; // 1/sqrt(1024)
constexpr size_t KVELEMS = (size_t)Bc * Hc * Tc * DHc;   // 4,194,304
constexpr size_t QBYTES  = KVELEMS * 2;                   // 8 MB per bf16 panel

// ---- WMMA fragment index helpers (wave32, CDNA5 ISA 7.12.2) ----
// A 16x32 (16-bit): lane L -> m = L&15; element i -> K = half*16 + i (contiguous k order used here; B matches)
__device__ __forceinline__ int a_kk(int half, int i) {
  return half * 16 + i;
}
// B 32x16 (16-bit): lane L -> n = L&15 ; element i -> K = (L>>4)*16 + i (same contiguous order as A here)
// C/D f32 16x16:    lane L, vgpr r -> m = r + (L>>4)*8 ; n = L&15

__device__ __forceinline__ float xor16_max(float x) {
#pragma unroll
  for (int m = 1; m < 16; m <<= 1) x = fmaxf(x, __shfl_xor(x, m, 32));
  return x;
}
__device__ __forceinline__ float xor16_sum(float x) {
#pragma unroll
  for (int m = 1; m < 16; m <<= 1) x += __shfl_xor(x, m, 32);
  return x;
}

// ---- TDM: 2-D tensor_load_to_lds, data_size = 2 bytes (D# per ISA 8.3/8.4) ----
// This toolchain exposes the 6-arg builtin:
//   (uint32x4 g0, int32x8 g1, int32x4 g2, int32x4 g3, int32x8 pad, i32 cpol)
__device__ __forceinline__ void tdm_load_2d_bf16(unsigned lds_off, const bf16_t* gptr,
                                                 unsigned tensor_d0, unsigned tensor_d1,
                                                 unsigned long long d0_stride,
                                                 unsigned tile_d0, unsigned tile_d1) {
  const unsigned long long ga = (unsigned long long)(uintptr_t)gptr;
  v4u g0;
  g0[0] = 1u;                                            // count=1, user descriptor
  g0[1] = lds_off;                                       // lds_addr (bytes)
  g0[2] = (unsigned)(ga & 0xffffffffu);                  // global_addr[31:0]
  g0[3] = (unsigned)((ga >> 32) & 0x01ffffffu) | (2u << 30); // addr[56:32] | type=2
  v8i g1;
  g1[0] = (int)(1u << 16);                               // data_size=2B; mask/flags 0
  g1[1] = (int)((tensor_d0 & 0xffffu) << 16);            // tensor_dim0[15:0] @ bit48
  g1[2] = (int)((tensor_d0 >> 16) | ((tensor_d1 & 0xffffu) << 16));
  g1[3] = (int)((tensor_d1 >> 16) | (tile_d0 << 16));    // tile_dim0 @ bit112
  g1[4] = (int)(tile_d1 & 0xffffu);                      // tile_dim1; tile_dim2=0
  g1[5] = (int)(d0_stride & 0xffffffffu);                // tensor_dim0_stride[31:0]
  g1[6] = (int)((d0_stride >> 32) & 0xffffu);            // stride[47:32]; dim1_stride=0
  g1[7] = 0;
  const v4i z4 = {0, 0, 0, 0};                           // 2-D: groups 2/3 unused
  const v8i z8 = {0, 0, 0, 0, 0, 0, 0, 0};
  __builtin_amdgcn_tensor_load_to_lds(g0, g1, z4, z4, z8, 0);
}

// ============================================================
// Kernel 0: K -> bf16 (same [B,H,T,DH] layout), V -> bf16 transposed [B,H,DH,T]
// ============================================================
__global__ __launch_bounds__(256) void convkv_kernel(const float* __restrict__ kg,
                                                     const float* __restrict__ vg,
                                                     bf16_t* __restrict__ kws,
                                                     bf16_t* __restrict__ vws) {
  const size_t idx = (size_t)blockIdx.x * 256 + threadIdx.x;   // < KVELEMS
  kws[idx] = (bf16_t)kg[idx];
  const size_t dh = idx & 63;
  const size_t t  = (idx >> 6) & (Tc - 1);
  const size_t bh = idx >> 17;                                  // /(T*DH)
  vws[(bh * DHc + dh) * Tc + t] = (bf16_t)vg[idx];
}

// ============================================================
// Kernel 1: q = x @ Wq^T -> bf16 workspace in [B,H,T,DH] layout
// One wave per 16x16 output tile; K-loop of 32 (bf16 WMMA, f32 accum).
// ============================================================
__global__ __launch_bounds__(256) void qproj_kernel(const float* __restrict__ x,
                                                    const float* __restrict__ Wq,
                                                    bf16_t* __restrict__ qws) {
  const int lane = threadIdx.x & 31;
  const int wid  = threadIdx.x >> 5;
  const int tile = blockIdx.x * 8 + wid;          // 16384 tiles total
  const int colTiles = Dc / 16;                   // 64
  const int t0 = (tile / colTiles) * 16;          // flat row over B*T
  const int e0 = (tile % colTiles) * 16;          // output feature
  const int half = lane >> 4;
  const int nl   = lane & 15;

  v8f c = {};
#pragma unroll 4
  for (int k0 = 0; k0 < Dc; k0 += 32) {
    v16bf a, b;
#pragma unroll
    for (int i = 0; i < 16; ++i)
      a[i] = (bf16_t)x[(size_t)(t0 + nl) * Dc + k0 + a_kk(half, i)];
#pragma unroll
    for (int i = 0; i < 16; ++i)                  // B(kk,n) = Wq[e0+n, k0+kk]
      b[i] = (bf16_t)Wq[(size_t)(e0 + nl) * Dc + k0 + half * 16 + i];
    c = __builtin_amdgcn_wmma_f32_16x16x32_bf16(false, a, false, b, (short)0, c,
                                                false, false);
  }

  const int bb = t0 / Tc;
  const int tq = t0 % Tc;
#pragma unroll
  for (int r = 0; r < 8; ++r) {
    const int m  = r + half * 8;
    const int e  = e0 + nl;
    const int h  = e >> 6;
    const int dh = e & 63;
    qws[(((size_t)bb * Hc + h) * Tc + (tq + m)) * DHc + dh] = (bf16_t)c[r];
  }
}

// ============================================================
// Kernel 2: one wave per (b,h,16-query) tile. TDM double-buffers 32x64 K
// and 64x32 V^T bf16 tiles into LDS (s_wait_tensorcnt pipeline); QK^T and
// PV via bf16 WMMA; online softmax with 16-lane xor reductions.
// ============================================================
__global__ __launch_bounds__(32) void attn_kernel(const bf16_t* __restrict__ qws,
                                                  const bf16_t* __restrict__ kws,
                                                  const bf16_t* __restrict__ vws,
                                                  const float* __restrict__ xg,
                                                  float* __restrict__ out) {
  // [buf0: K 4KB | V 4KB][buf1: K 4KB | V 4KB][P 16x32 bf16 1KB]
  __shared__ unsigned char smem[2 * 8192 + 1024];
  bf16_t* const pstage = (bf16_t*)(smem + 16384);

  const int lane = threadIdx.x & 31;
  const int tile = blockIdx.x;                    // 4096 tiles
  const int mTiles = Tc / 16;                     // 128
  const int bh = tile / mTiles;
  const int m0 = (tile % mTiles) * 16;
  const int bb = bh / Hc, hh = bh % Hc;
  const size_t kvBase = (size_t)bh * Tc * DHc;    // also base of [DH,T] V panel
  const int half = lane >> 4;
  const int nl   = lane & 15;

  // Q fragments: two dh-chunks of 32 (bf16 direct from workspace)
  v16bf qa0, qa1;
#pragma unroll
  for (int i = 0; i < 16; ++i) {
    const size_t qrow = kvBase + (size_t)(m0 + nl) * DHc;
    qa0[i] = qws[qrow + a_kk(half, i)];
    qa1[i] = qws[qrow + 32 + a_kk(half, i)];
  }

  v8f acc0 = {}, acc1 = {}, acc2 = {}, acc3 = {};
  float row_m[8], row_l[8];
#pragma unroll
  for (int r = 0; r < 8; ++r) { row_m[r] = -__builtin_inff(); row_l[r] = 0.0f; }

  // prime the pipeline: buffer 0 <- key block 0
  tdm_load_2d_bf16(0,    kws + kvBase, DHc, Tc, DHc, /*tile*/ DHc, 32);   // K 32x64
  tdm_load_2d_bf16(4096, vws + kvBase, Tc, DHc, Tc,  /*tile*/ 32,  DHc);  // V^T 64x32

  const int kend = m0 + 16;
  int cur = 0;
  for (int kb = 0; kb < kend; kb += 32) {
    const bool more = (kb + 32) < kend;
    if (more) {
      // recycle buffer cur^1: make sure our old DS reads from it retired
      asm volatile("s_wait_dscnt 0x0" ::: "memory");
      const unsigned boff = (unsigned)((cur ^ 1) * 8192);
      tdm_load_2d_bf16(boff,        kws + kvBase + (size_t)(kb + 32) * DHc,
                       DHc, Tc, DHc, DHc, 32);
      tdm_load_2d_bf16(boff + 4096, vws + kvBase + (size_t)(kb + 32),
                       Tc, DHc, Tc, 32, DHc);
      __builtin_amdgcn_s_wait_tensorcnt(2);       // current buffer's pair done
    } else {
      __builtin_amdgcn_s_wait_tensorcnt(0);
    }
    asm volatile("" ::: "memory");                // don't hoist LDS reads above wait

    const bf16_t* ks = (const bf16_t*)(smem + cur * 8192);         // [key][dh]
    const bf16_t* vs = (const bf16_t*)(smem + cur * 8192 + 4096);  // [dh][key]

    // ---- scores S = Q @ K^T for keys [kb, kb+32) ----
    v16bf kb00, kb01, kb10, kb11;
#pragma unroll
    for (int i = 0; i < 16; ++i) {
      const int kk = half * 16 + i;               // dh within 32-chunk
      kb00[i] = ks[nl * 64 + kk];
      kb01[i] = ks[nl * 64 + 32 + kk];
      kb10[i] = ks[(16 + nl) * 64 + kk];
      kb11[i] = ks[(16 + nl) * 64 + 32 + kk];
    }
    v8f s0 = {}, s1 = {};
    s0 = __builtin_amdgcn_wmma_f32_16x16x32_bf16(false, qa0, false, kb00, (short)0, s0, false, false);
    s0 = __builtin_amdgcn_wmma_f32_16x16x32_bf16(false, qa1, false, kb01, (short)0, s0, false, false);
    s1 = __builtin_amdgcn_wmma_f32_16x16x32_bf16(false, qa0, false, kb10, (short)0, s1, false, false);
    s1 = __builtin_amdgcn_wmma_f32_16x16x32_bf16(false, qa1, false, kb11, (short)0, s1, false, false);

    // ---- causal mask + online softmax ----
#pragma unroll
    for (int r = 0; r < 8; ++r) {
      const int qi = m0 + r + half * 8;
      const float v0 = (kb + nl      <= qi) ? s0[r] * SCALE : -__builtin_inff();
      const float v1 = (kb + 16 + nl <= qi) ? s1[r] * SCALE : -__builtin_inff();
      float mx = xor16_max(fmaxf(v0, v1));
      mx = fmaxf(mx, row_m[r]);
      const float alpha = __expf(row_m[r] - mx);  // 0 on first block
      const float p0 = __expf(v0 - mx);
      const float p1 = __expf(v1 - mx);
      row_l[r] = row_l[r] * alpha + xor16_sum(p0 + p1);
      row_m[r] = mx;
      acc0[r] *= alpha; acc1[r] *= alpha; acc2[r] *= alpha; acc3[r] *= alpha;
      pstage[(r + half * 8) * 32 + nl]      = (bf16_t)p0;  // C-layout -> row-major
      pstage[(r + half * 8) * 32 + 16 + nl] = (bf16_t)p1;
    }
    asm volatile("s_wait_dscnt 0x0" ::: "memory"); // order intra-wave P RAW

    v16bf pa;                                      // P as A fragment (16x32)
#pragma unroll
    for (int i = 0; i < 16; ++i)
      pa[i] = pstage[nl * 32 + a_kk(half, i)];

    // ---- O += P @ V : four 16-wide dh N-tiles ----
    v8f* accs[4] = { &acc0, &acc1, &acc2, &acc3 };
#pragma unroll
    for (int t = 0; t < 4; ++t) {
      v16bf vb;
#pragma unroll
      for (int i = 0; i < 16; ++i)                 // B(kk,n): key=kb+kk, dh=t*16+n
        vb[i] = vs[(t * 16 + nl) * 32 + half * 16 + i];
      *accs[t] = __builtin_amdgcn_wmma_f32_16x16x32_bf16(false, pa, false, vb,
                                                         (short)0, *accs[t], false, false);
    }
    cur ^= 1;
  }

  // ---- epilogue: normalize + residual ----
#pragma unroll
  for (int r = 0; r < 8; ++r) {
    const int tq = m0 + r + half * 8;
    const float inv = 1.0f / row_l[r];
    const size_t base = ((size_t)bb * Tc + tq) * Dc + hh * DHc;
    out[base +  0 + nl] = xg[base +  0 + nl] + acc0[r] * inv;
    out[base + 16 + nl] = xg[base + 16 + nl] + acc1[r] * inv;
    out[base + 32 + nl] = xg[base + 32 + nl] + acc2[r] * inv;
    out[base + 48 + nl] = xg[base + 48 + nl] + acc3[r] * inv;
  }
}

extern "C" void kernel_launch(void* const* d_in, const int* in_sizes, int n_in,
                              void* d_out, int out_size, void* d_ws, size_t ws_size,
                              hipStream_t stream) {
  const float* x  = (const float*)d_in[0];
  const float* k  = (const float*)d_in[1];
  const float* v  = (const float*)d_in[2];
  const float* Wq = (const float*)d_in[3];
  float* out = (float*)d_out;
  bf16_t* qws = (bf16_t*)d_ws;                              // 8 MB
  bf16_t* kws = (bf16_t*)((char*)d_ws + QBYTES);            // 8 MB
  bf16_t* vws = (bf16_t*)((char*)d_ws + 2 * QBYTES);        // 8 MB (transposed)

  convkv_kernel<<<(int)(KVELEMS / 256), 256, 0, stream>>>(k, v, kws, vws);
  qproj_kernel<<<2048, 256, 0, stream>>>(x, Wq, qws);       // 16384 wave-tiles
  attn_kernel<<<4096, 32, 0, stream>>>(qws, kws, vws, x, out);
}
